// MoeBlock_11519102288545
// MI455X (gfx1250) — matrix operand order repaired and checked by reading, not compiled, not run on hardware
//
#include <hip/hip_runtime.h>
#include <hip/hip_bf16.h>

// ---------------- problem constants (from reference) ----------------
#define T_TOK 4096   // B*L
#define EDIM  1024   // model dim E
#define HDIM  2048   // hidden dim H
#define NEXP  8      // experts
#define ROWT  128    // T_TOK / MT row tiles per expert (worst case)
#define MT    32     // rows per block tile
#define NT    128    // output cols per block tile

// LDS strides (bf16 units), padded for 64-bank LDS, 16B-aligned rows
#define AS_STRIDE 40    // 80 B  (bank step 20 words -> 16 distinct banks)
#define BS_STRIDE 136   // 272 B (bank step 68 mod 64 = 4 words)
#define A_TILE (MT * AS_STRIDE)    // ushorts per 32-K A sub-tile
#define B_TILE (32 * BS_STRIDE)    // ushorts per 32-K B sub-tile

// workspace layout (bytes)
#define WS_CNT_OFF   0            // 8 ints (zeroed each launch, 256 B reserved)
#define WS_ENT_OFF   256          // NEXP*T_TOK ints = 128 KB
#define WS_ROWW_OFF  (256 + NEXP*T_TOK*4)  // 8192 floats
#define WS_INTER_OFF (256*1024)   // bf16 intermediate: 8192 x HDIM = 32 MB

typedef __attribute__((ext_vector_type(16))) __bf16 v16bf;
typedef __attribute__((ext_vector_type(8)))  __bf16 v8bf;
typedef __attribute__((ext_vector_type(2)))  __bf16 v2bf;
typedef __attribute__((ext_vector_type(8)))  float  v8f;

#if defined(__has_builtin)
#if __has_builtin(__builtin_amdgcn_cvt_pk_bf16_f32)
#define HAVE_PK_BF16 1
#endif
#endif

static __device__ __forceinline__ unsigned short f2b(float f) {
    unsigned int u = __float_as_uint(f);
    u += 0x7FFFu + ((u >> 16) & 1u);      // round to nearest even
    return (unsigned short)(u >> 16);
}

// Pack two f32 -> two bf16 in one dword.
// Fallback: round-half-up (+0x8000) then one v_perm_b32 grabs both high
// halves -> 3 VALU per 2 elements (tie-breaking differs from RNE only on
// exact ties, far below bf16 quantization noise).
static __device__ __forceinline__ unsigned int f2x2(float a, float b) {
#ifdef HAVE_PK_BF16
    v2bf v = __builtin_amdgcn_cvt_pk_bf16_f32(a, b);
    return __builtin_bit_cast(unsigned int, v);
#else
    const unsigned int ua = __float_as_uint(a) + 0x8000u;
    const unsigned int ub = __float_as_uint(b) + 0x8000u;
    // result bytes: {ub.b3, ub.b2, ua.b3, ua.b2} = (hi16(b)<<16)|hi16(a)
    return __builtin_amdgcn_perm(ub, ua, 0x07060302u);
#endif
}

static __device__ __forceinline__ uint2 cvt4(float4 v) {
    uint2 r;
    r.x = f2x2(v.x, v.y);
    r.y = f2x2(v.z, v.w);
    return r;
}

static __device__ __forceinline__ v16bf make_frag(const unsigned short* p0,
                                                  const unsigned short* p1) {
    v8bf lo = *(const v8bf*)p0;
    v8bf hi = *(const v8bf*)p1;
    v16bf r;
#pragma unroll
    for (int i = 0; i < 8; ++i) { r[i] = lo[i]; r[i + 8] = hi[i]; }
    return r;
}

// A fragment (16x32 bf16): lane<16 holds row (lane), K = {0..7, 16..23};
// lane>=16 holds K = {8..15, 24..31}.  (ISA 7.12.2 table)
static __device__ __forceinline__ v16bf load_afrag(const unsigned short* aS,
                                                   int mBase, int laneMod, int laneHalf) {
    const unsigned short* p = aS + (mBase + laneMod) * AS_STRIDE + laneHalf * 8;
    return make_frag(p, p + 16);
}

// B fragment (32x16 bf16): lane holds column N = lane%16; lane<16 holds
// K = {0..7, 16..23}, lane>=16 holds K = {8..15, 24..31} (matches A).
static __device__ __forceinline__ v16bf load_bfrag(const unsigned short* bS,
                                                   int laneMod, int laneHalf, int nOff) {
    const unsigned short* p = bS + (laneHalf * 8) * BS_STRIDE + nOff + laneMod;
    v16bf r;
#pragma unroll
    for (int i = 0; i < 8; ++i) {
        r[i]     = __builtin_bit_cast(__bf16, p[i * BS_STRIDE]);
        r[i + 8] = __builtin_bit_cast(__bf16, p[(16 + i) * BS_STRIDE]);
    }
    return r;
}

static __device__ __forceinline__ v8f wmma_bf16(v16bf a, v16bf b, v8f c) {
    return __builtin_amdgcn_wmma_f32_16x16x32_bf16(
        /*neg_a=*/false, a, /*neg_b=*/false, b,
        /*c_mod=*/(short)0, c, /*reuse_a=*/false, /*reuse_b=*/false);
}

// ---------------------------------------------------------------------------
// Kernel 1: router.  One wave32 per token: 8 gate logits, top-2, softmax,
// build per-expert gathered row lists (row id = tok*2 + k) + row weights.
// ---------------------------------------------------------------------------
__global__ void moe_router_kernel(const float* __restrict__ x,
                                  const float* __restrict__ gate,
                                  int* __restrict__ cnt,
                                  int* __restrict__ ent,
                                  float* __restrict__ rowW) {
    const int wave = threadIdx.x >> 5;
    const int lane = threadIdx.x & 31;
    const int t = blockIdx.x * 8 + wave;

    float acc[NEXP];
#pragma unroll
    for (int n = 0; n < NEXP; ++n) acc[n] = 0.f;

    const float* xr = x + (size_t)t * EDIM;
#pragma unroll 4
    for (int i = 0; i < EDIM / 32; ++i) {
        const int e = lane + i * 32;
        const float xv = xr[e];
        const float4* gp = (const float4*)(gate + (size_t)e * NEXP);
        const float4 g0 = gp[0];
        const float4 g1 = gp[1];
        acc[0] = fmaf(xv, g0.x, acc[0]);
        acc[1] = fmaf(xv, g0.y, acc[1]);
        acc[2] = fmaf(xv, g0.z, acc[2]);
        acc[3] = fmaf(xv, g0.w, acc[3]);
        acc[4] = fmaf(xv, g1.x, acc[4]);
        acc[5] = fmaf(xv, g1.y, acc[5]);
        acc[6] = fmaf(xv, g1.z, acc[6]);
        acc[7] = fmaf(xv, g1.w, acc[7]);
    }
#pragma unroll
    for (int n = 0; n < NEXP; ++n) {
#pragma unroll
        for (int off = 16; off > 0; off >>= 1)
            acc[n] += __shfl_xor(acc[n], off, 32);
    }

    if (lane == 0) {
        int i0 = 0; float v0 = acc[0];
#pragma unroll
        for (int n = 1; n < NEXP; ++n)
            if (acc[n] > v0) { v0 = acc[n]; i0 = n; }
        int i1 = (i0 == 0) ? 1 : 0; float v1 = acc[i1];
#pragma unroll
        for (int n = 0; n < NEXP; ++n)
            if (n != i0 && acc[n] > v1) { v1 = acc[n]; i1 = n; }
        // softmax over {v0, v1}, v0 >= v1
        const float s1 = __expf(v1 - v0);
        const float inv = 1.f / (1.f + s1);
        const int r0 = atomicAdd(&cnt[i0], 1);
        ent[i0 * T_TOK + r0] = t * 2 + 0;
        rowW[t * 2 + 0] = inv;
        const int r1 = atomicAdd(&cnt[i1], 1);
        ent[i1 * T_TOK + r1] = t * 2 + 1;
        rowW[t * 2 + 1] = s1 * inv;
    }
}

// ---------------------------------------------------------------------------
// Kernel 2: grouped GEMM, FFN in.  inter[row] = bf16(silu(X W0) * (X W1)),
// rows gathered per expert list.  Double-buffered LDS, K=64 per stage
// (two 32-K sub-chunks): 1 barrier per 8 WMMAs.
// 256 threads = 8 waves; block tile 32x128; each wave: 16x32 output.
// ---------------------------------------------------------------------------
__global__ void __launch_bounds__(256)
moe_ffn1_kernel(const float* __restrict__ x,
                const float* __restrict__ w0,
                const float* __restrict__ w1,
                const int* __restrict__ cnt,
                const int* __restrict__ ent,
                unsigned short* __restrict__ inter) {
    const int e  = blockIdx.y >> 7;     // blockIdx.y = e*ROWT + rt
    const int rt = blockIdx.y & (ROWT - 1);
    const int cnt_e = cnt[e];
    const int rowBase = rt * MT;
    if (rowBase >= cnt_e) return;       // uniform early exit (before barriers)

    // 2 pipeline buffers x 2 K-sub-chunks each
    __shared__ __align__(16) unsigned short aS[4 * A_TILE];
    __shared__ __align__(16) unsigned short b0S[4 * B_TILE];
    __shared__ __align__(16) unsigned short b1S[4 * B_TILE];
    __shared__ int rowValS[MT];

    const int tid = threadIdx.x;
    if (tid < MT) {
        const int idx = rowBase + tid;
        rowValS[tid] = (idx < cnt_e) ? ent[e * T_TOK + idx] : 0;
    }

    const int nBase = blockIdx.x * NT;
    const float* w0b = w0 + (size_t)e * EDIM * HDIM;
    const float* w1b = w1 + (size_t)e * EDIM * HDIM;

    const int lane = tid & 31;
    const int laneMod = lane & 15;
    const int laneHalf = lane >> 4;
    const int wave = tid >> 5;
    const int mBase = (wave >> 2) * 16;
    const int nOff0 = (wave & 3) * 32;

    const int ar = tid >> 3, ac = tid & 7;          // A: 32 rows x 8 float4

    v8f c00 = {}, c01 = {}, c10 = {}, c11 = {};

    // substage: one 32-K chunk (global fp32 -> bf16 -> LDS slot)
    auto substage = [&](int k0, int slot) {
        unsigned short* aB  = aS  + slot * A_TILE;
        unsigned short* b0B = b0S + slot * B_TILE;
        unsigned short* b1B = b1S + slot * B_TILE;
        {
            const int tok = rowValS[ar] >> 1;
            const float4 xv = *(const float4*)(x + (size_t)tok * EDIM + k0 + ac * 4);
            *(uint2*)&aB[ar * AS_STRIDE + ac * 4] = cvt4(xv);
        }
#pragma unroll
        for (int j = 0; j < 4; ++j) {
            const int f  = tid + j * 256;
            const int kk = f >> 5;
            const int c4 = f & 31;
            const size_t go = (size_t)(k0 + kk) * HDIM + nBase + c4 * 4;
            const float4 a0 = *(const float4*)(w0b + go);
            const float4 a1 = *(const float4*)(w1b + go);
            *(uint2*)&b0B[kk * BS_STRIDE + c4 * 4] = cvt4(a0);
            *(uint2*)&b1B[kk * BS_STRIDE + c4 * 4] = cvt4(a1);
        }
    };

    substage(0, 0);
    substage(32, 1);
    __syncthreads();

    const int NKK = EDIM / 64;
    for (int i = 0; i < NKK; ++i) {
        const int par = i & 1;
        if (i + 1 < NKK) {                           // prefetch next 64-K stage
            substage((i + 1) * 64,      (par ^ 1) * 2);
            substage((i + 1) * 64 + 32, (par ^ 1) * 2 + 1);
        }
#pragma unroll
        for (int cch = 0; cch < 2; ++cch) {
            const int slot = par * 2 + cch;
            const unsigned short* aB  = aS  + slot * A_TILE;
            const unsigned short* b0B = b0S + slot * B_TILE;
            const unsigned short* b1B = b1S + slot * B_TILE;
            const v16bf a = load_afrag(aB, mBase, laneMod, laneHalf);
            c00 = wmma_bf16(a, load_bfrag(b0B, laneMod, laneHalf, nOff0),      c00);
            c01 = wmma_bf16(a, load_bfrag(b0B, laneMod, laneHalf, nOff0 + 16), c01);
            c10 = wmma_bf16(a, load_bfrag(b1B, laneMod, laneHalf, nOff0),      c10);
            c11 = wmma_bf16(a, load_bfrag(b1B, laneMod, laneHalf, nOff0 + 16), c11);
        }
        __syncthreads();
    }

    // ---- epilogue: silu(w0)*w1 (fast rcp sigmoid), bf16 store ----
#pragma unroll
    for (int s = 0; s < 2; ++s) {
        const v8f ca = s ? c01 : c00;
        const v8f cb = s ? c11 : c10;
        const int col = nBase + nOff0 + s * 16 + laneMod;
#pragma unroll
        for (int r = 0; r < 8; ++r) {
            const int m = mBase + laneHalf * 8 + r;
            if (rowBase + m < cnt_e) {
                const float v0 = ca[r];
                const float h = v0 * __builtin_amdgcn_rcpf(1.f + __expf(-v0)) * cb[r];
                inter[(size_t)rowValS[m] * HDIM + col] = f2b(h);
            }
        }
    }
}

// ---------------------------------------------------------------------------
// Kernel 3: grouped GEMM, FFN out.  out[tok] += p * (inter_row x Wo_e).
// ---------------------------------------------------------------------------
__global__ void __launch_bounds__(256)
moe_ffn2_kernel(const unsigned short* __restrict__ inter,
                const float* __restrict__ wo,
                const int* __restrict__ cnt,
                const int* __restrict__ ent,
                const float* __restrict__ rowW,
                float* __restrict__ out) {
    const int e  = blockIdx.y >> 7;
    const int rt = blockIdx.y & (ROWT - 1);
    const int cnt_e = cnt[e];
    const int rowBase = rt * MT;
    if (rowBase >= cnt_e) return;

    __shared__ __align__(16) unsigned short aS[4 * A_TILE];
    __shared__ __align__(16) unsigned short bS[4 * B_TILE];
    __shared__ int rowValS[MT];
    __shared__ float wS[MT];

    const int tid = threadIdx.x;
    if (tid < MT) {
        const int idx = rowBase + tid;
        int v = 0; float w = 0.f;
        if (idx < cnt_e) { v = ent[e * T_TOK + idx]; w = rowW[v]; }
        rowValS[tid] = v;
        wS[tid] = w;
    }

    const int nBase = blockIdx.x * NT;
    const float* wob = wo + (size_t)e * HDIM * EDIM;

    const int lane = tid & 31;
    const int laneMod = lane & 15;
    const int laneHalf = lane >> 4;
    const int wave = tid >> 5;
    const int mBase = (wave >> 2) * 16;
    const int nOff0 = (wave & 3) * 32;

    const int ar = tid >> 2, aq = tid & 3;          // A: 32 rows x 4 uint4

    v8f c0 = {}, c1 = {};

    auto substage = [&](int k0, int slot) {
        unsigned short* aB = aS + slot * A_TILE;
        unsigned short* bB = bS + slot * B_TILE;
        if (tid < 128) {                            // bf16 A rows: direct copy
            const uint4 v = *(const uint4*)(inter + (size_t)rowValS[ar] * HDIM + k0 + aq * 8);
            *(uint4*)&aB[ar * AS_STRIDE + aq * 8] = v;
        }
#pragma unroll
        for (int j = 0; j < 4; ++j) {
            const int f  = tid + j * 256;
            const int kk = f >> 5;
            const int c4 = f & 31;
            const size_t go = (size_t)(k0 + kk) * EDIM + nBase + c4 * 4;
            const float4 a0 = *(const float4*)(wob + go);
            *(uint2*)&bB[kk * BS_STRIDE + c4 * 4] = cvt4(a0);
        }
    };

    substage(0, 0);
    substage(32, 1);
    __syncthreads();

    const int NKK = HDIM / 64;
    for (int i = 0; i < NKK; ++i) {
        const int par = i & 1;
        if (i + 1 < NKK) {
            substage((i + 1) * 64,      (par ^ 1) * 2);
            substage((i + 1) * 64 + 32, (par ^ 1) * 2 + 1);
        }
#pragma unroll
        for (int cch = 0; cch < 2; ++cch) {
            const int slot = par * 2 + cch;
            const unsigned short* aB = aS + slot * A_TILE;
            const unsigned short* bB = bS + slot * B_TILE;
            const v16bf a = load_afrag(aB, mBase, laneMod, laneHalf);
            c0 = wmma_bf16(a, load_bfrag(bB, laneMod, laneHalf, nOff0),      c0);
            c1 = wmma_bf16(a, load_bfrag(bB, laneMod, laneHalf, nOff0 + 16), c1);
        }
        __syncthreads();
    }

    // ---- epilogue: scale by routing prob, atomic scatter-add into out ----
#pragma unroll
    for (int s = 0; s < 2; ++s) {
        const v8f cc = s ? c1 : c0;
        const int col = nBase + nOff0 + s * 16 + laneMod;
#pragma unroll
        for (int r = 0; r < 8; ++r) {
            const int m = mBase + laneHalf * 8 + r;
            if (rowBase + m < cnt_e) {
                const int tok = rowValS[m] >> 1;
                atomicAdd(&out[(size_t)tok * EDIM + col], cc[r] * wS[m]);
            }
        }
    }
}

// ---------------------------------------------------------------------------
extern "C" void kernel_launch(void* const* d_in, const int* in_sizes, int n_in,
                              void* d_out, int out_size, void* d_ws, size_t ws_size,
                              hipStream_t stream) {
    const float* x    = (const float*)d_in[0];   // [2,2048,1024]
    const float* gate = (const float*)d_in[1];   // [1024,8]
    const float* w0   = (const float*)d_in[2];   // [8,1024,2048]
    const float* w1   = (const float*)d_in[3];   // [8,1024,2048]
    const float* wo   = (const float*)d_in[4];   // [8,2048,1024]
    float* out = (float*)d_out;                  // [2,2048,1024]

    char* ws = (char*)d_ws;                      // needs ~34 MB
    int*   cnt  = (int*)(ws + WS_CNT_OFF);
    int*   ent  = (int*)(ws + WS_ENT_OFF);
    float* rowW = (float*)(ws + WS_ROWW_OFF);
    unsigned short* inter = (unsigned short*)(ws + WS_INTER_OFF);

    hipMemsetAsync(cnt, 0, 256, stream);
    hipMemsetAsync(out, 0, (size_t)out_size * sizeof(float), stream);

    moe_router_kernel<<<T_TOK / 8, 256, 0, stream>>>(x, gate, cnt, ent, rowW);
    moe_ffn1_kernel<<<dim3(HDIM / NT, NEXP * ROWT), 256, 0, stream>>>(
        x, w0, w1, cnt, ent, inter);
    moe_ffn2_kernel<<<dim3(EDIM / NT, NEXP * ROWT), 256, 0, stream>>>(
        inter, wo, cnt, ent, rowW, out);
}
